// SACB_27410481283469
// MI455X (gfx1250) — matrix operand order repaired and checked by reading, not compiled, not run on hardware
//
#include <hip/hip_runtime.h>

typedef __attribute__((ext_vector_type(16))) __bf16        v16bf;
typedef __attribute__((ext_vector_type(8)))  float         v8f;
typedef __attribute__((ext_vector_type(8)))  unsigned int  v8u;

#define NSP   64000   // 40^3 spatial positions
#define CCH   64
#define OCH   64
#define DD    40
#define K3    27
#define NUMK  4
#define FK    1728    // C*K3
#define KSTEPS 54     // FK / 32
#define NT    48      // padded N-tile (40 valid cols -> 3 x 16)
#define ROWPITCH 1736 // FK + 8 pad (ushorts); row stride 3472B, 16B aligned

__device__ __forceinline__ unsigned short f2bf(float f) {
    unsigned u = __float_as_uint(f);
    unsigned r = u + 0x7FFFu + ((u >> 16) & 1u); // RNE
    return (unsigned short)(r >> 16);
}

__device__ __forceinline__ v16bf mkbf16(uint4 lo, uint4 hi) {
    v8u u;
    u[0] = lo.x; u[1] = lo.y; u[2] = lo.z; u[3] = lo.w;
    u[4] = hi.x; u[5] = hi.y; u[6] = hi.z; u[7] = hi.w;
    return __builtin_bit_cast(v16bf, u);
}

// ---------------- stage 1: channel projection xp = proj_w @ x ----------------
__global__ void k_proj(const float* __restrict__ x, const float* __restrict__ pw,
                       float* __restrict__ xp) {
    long g = (long)blockIdx.x * blockDim.x + threadIdx.x;
    if (g >= (long)8 * NSP) return;
    int n = (int)(g % NSP);
    int og = (int)(g / NSP);            // group of 8 output channels
    float acc[8] = {0.f,0.f,0.f,0.f,0.f,0.f,0.f,0.f};
    for (int i = 0; i < CCH; ++i) {
        float xv = x[(long)i * NSP + n];
#pragma unroll
        for (int j = 0; j < 8; ++j)
            acc[j] += pw[(og * 8 + j) * CCH + i] * xv;
    }
#pragma unroll
    for (int j = 0; j < 8; ++j)
        xp[(long)(og * 8 + j) * NSP + n] = acc[j];
}

// ---------------- stage 2: 27-tap zero-padded box mean -> kmeans points -----
__global__ void k_mean(const float* __restrict__ xp, float* __restrict__ pts) {
    long g = (long)blockIdx.x * blockDim.x + threadIdx.x;
    if (g >= (long)CCH * NSP) return;
    int n = (int)(g % NSP);
    int c = (int)(g / NSP);
    int z = n / 1600, r = n % 1600, y = r / DD, x = r % DD;
    float s = 0.f;
    for (int dz = -1; dz <= 1; ++dz)
        for (int dy = -1; dy <= 1; ++dy)
            for (int dx = -1; dx <= 1; ++dx) {
                int zz = z + dz, yy = y + dy, xx = x + dx;
                if ((unsigned)zz < DD && (unsigned)yy < DD && (unsigned)xx < DD)
                    s += xp[(long)c * NSP + zz * 1600 + yy * DD + xx];
            }
    pts[g] = s * (1.f / 27.f);
}

// ---------------- kmeans ----------------
__global__ void k_initcent(const float* __restrict__ pts, float* __restrict__ cent) {
    int t = threadIdx.x;
    if (t < NUMK * CCH) {
        int k = t / CCH, c = t % CCH;
        cent[t] = pts[(long)c * NSP + k];     // cent0 = pts[:4]
    }
}

__global__ void k_assign(const float* __restrict__ pts, const float* __restrict__ cent,
                         int* __restrict__ idx) {
    __shared__ float cs[NUMK * CCH];
    int tid = threadIdx.x;
    if (tid < NUMK * CCH) cs[tid] = cent[tid];
    __syncthreads();
    int n = blockIdx.x * blockDim.x + tid;
    if (n >= NSP) return;
    float d0 = 0.f, d1 = 0.f, d2 = 0.f, d3 = 0.f;
    for (int c = 0; c < CCH; ++c) {
        float p = pts[(long)c * NSP + n];
        float t0 = p - cs[0 * CCH + c]; d0 += t0 * t0;
        float t1 = p - cs[1 * CCH + c]; d1 += t1 * t1;
        float t2 = p - cs[2 * CCH + c]; d2 += t2 * t2;
        float t3 = p - cs[3 * CCH + c]; d3 += t3 * t3;
    }
    int best = 0; float bd = d0;
    if (d1 < bd) { bd = d1; best = 1; }
    if (d2 < bd) { bd = d2; best = 2; }
    if (d3 < bd) { bd = d3; best = 3; }
    idx[n] = best;
}

// deterministic (atomic-free) cluster sum: one block per (k, c)
__global__ void k_reduce(const float* __restrict__ pts, const int* __restrict__ idx,
                         float* __restrict__ sums, float* __restrict__ cnts) {
    int k = blockIdx.x / CCH;
    int c = blockIdx.x % CCH;
    __shared__ float sm[256];
    __shared__ float sc[256];
    int tid = threadIdx.x;
    float s = 0.f, cn = 0.f;
    for (int n = tid; n < NSP; n += 256) {
        if (idx[n] == k) { s += pts[(long)c * NSP + n]; cn += 1.f; }
    }
    sm[tid] = s; sc[tid] = cn;
    __syncthreads();
    for (int off = 128; off > 0; off >>= 1) {
        if (tid < off) { sm[tid] += sm[tid + off]; sc[tid] += sc[tid + off]; }
        __syncthreads();
    }
    if (tid == 0) {
        sums[k * CCH + c] = sm[0];
        if (c == 0) cnts[k] = sc[0];
    }
}

__global__ void k_update(const float* __restrict__ sums, const float* __restrict__ cnts,
                         float* __restrict__ cent) {
    int t = threadIdx.x;
    if (t < NUMK * CCH) cent[t] = sums[t] / fmaxf(cnts[t / CCH], 1.f);
}

// ---------------- gate MLPs (one workgroup) ----------------
__global__ void k_mlp(const float* __restrict__ cent,
                      const float* __restrict__ kw1, const float* __restrict__ kb1,
                      const float* __restrict__ kw2, const float* __restrict__ kb2,
                      const float* __restrict__ kw3, const float* __restrict__ kb3,
                      const float* __restrict__ bw1, const float* __restrict__ bb1,
                      const float* __restrict__ bw2, const float* __restrict__ bb2,
                      const float* __restrict__ bw3, const float* __restrict__ bb3,
                      float* __restrict__ wi, float* __restrict__ bg) {
    __shared__ float feat[256], h1[128], h2[128];
    int t = threadIdx.x;
    feat[t] = cent[t];                         // feat = cent.reshape(NUM_K*C)
    __syncthreads();
    if (t < 128) { float s = kb1[t]; for (int i = 0; i < 256; ++i) s += feat[i] * kw1[i * 128 + t]; h1[t] = fmaxf(s, 0.f); }
    __syncthreads();
    if (t < 128) { float s = kb2[t]; for (int i = 0; i < 128; ++i) s += h1[i] * kw2[i * 128 + t]; h2[t] = fmaxf(s, 0.f); }
    __syncthreads();
    if (t < NUMK * K3) { float s = kb3[t]; for (int i = 0; i < 128; ++i) s += h2[i] * kw3[i * (NUMK * K3) + t]; wi[t] = 1.f / (1.f + __expf(-s)); }
    __syncthreads();
    if (t < 64) { float s = bb1[t]; for (int i = 0; i < 256; ++i) s += feat[i] * bw1[i * 64 + t]; h1[t] = fmaxf(s, 0.f); }
    __syncthreads();
    if (t < 64) { float s = bb2[t]; for (int i = 0; i < 64; ++i) s += h1[i] * bw2[i * 64 + t]; h2[t] = fmaxf(s, 0.f); }
    __syncthreads();
    { float s = bb3[t]; for (int i = 0; i < 64; ++i) s += h2[i] * bw3[i * 256 + t]; bg[t] = s; }
}

// ---------------- pre-swizzle base_w into per-lane WMMA A-fragments ----------
// layout: [m(4)][s(54)][lane(32)][e(16)] bf16; lane<16: K=e, lane>=16: K=e+16
__global__ void k_packA(const float* __restrict__ base_w, unsigned short* __restrict__ pa) {
    int g = blockIdx.x * blockDim.x + threadIdx.x;
    if (g >= 4 * KSTEPS * 32 * 16) return;
    int e    = g & 15;
    int lane = (g >> 4) & 31;
    int s    = (g >> 9) % KSTEPS;
    int m    = g / (KSTEPS * 512);
    int o = m * 16 + (lane & 15);
    int K = (lane < 16) ? e : e + 16;
    int f = s * 32 + K;
    int c = f / K3, t = f % K3;
    pa[g] = f2bf(base_w[(o * CCH + c) * K3 + t]);
}

// ---------------- main fused GEMM: one workgroup per (z,y) row --------------
// 256 threads = 8 waves. Staging uses all 256 threads; the 12 WMMA tiles
// (4 M x 3 N) split wave-uniformly: waves 0-3 -> (m, j=0,1), waves 4-7 -> (m, j=2).
__global__ __launch_bounds__(256) void k_gemm(
    const float* __restrict__ xp, const unsigned short* __restrict__ pa,
    const int* __restrict__ idx, const float* __restrict__ wi,
    const float* __restrict__ bg, const float* __restrict__ xin,
    const float* __restrict__ prelu, float* __restrict__ out) {
    extern __shared__ unsigned short Xs[];      // [NT][ROWPITCH] bf16
    __shared__ int   kid[DD];
    __shared__ float wis[NUMK * K3];

    int tid = threadIdx.x;
    int zy = blockIdx.x;
    int z = zy / DD, y = zy % DD;
    int nb = zy * DD;                           // n = z*1600 + y*40 + x

    if (tid < NUMK * K3) wis[tid] = wi[tid];
    if (tid < DD) kid[tid] = idx[nb + tid];
    __syncthreads();

    // zero the 8 padding columns with wide stores: 8 rows * 3456B = 1728 uint4
    for (int i = tid; i < 8 * (FK / 8); i += 256) {   // 1728 = 256 * 6.75 -> guard-free? no
        int rr  = i / (FK / 8);                       // 0..7
        int off = i % (FK / 8);                       // 0..215
        *(uint4*)(&Xs[(DD + rr) * ROWPITCH + off * 8]) = make_uint4(0u, 0u, 0u, 0u);
    }

    // stage gate-scaled patch tile (valid cols only): 40*1728 = 69120 = 270*256
    for (int i = tid; i < DD * FK; i += 256) {
        int col = i % DD;
        int f   = i / DD;
        int c = f / K3, t = f % K3;
        int dz = t / 9, dy = (t / 3) % 3, dx = t % 3;
        int zz = z + dz - 1, yy = y + dy - 1, xx = col + dx - 1;
        float v = 0.f;
        if ((unsigned)zz < DD && (unsigned)yy < DD && (unsigned)xx < DD)
            v = wis[kid[col] * K3 + t] * xp[(long)c * NSP + zz * 1600 + yy * DD + xx];
        Xs[col * ROWPITCH + f] = f2bf(v);
    }
    __syncthreads();

    int wave = tid >> 5;                        // 0..7
    int lane = tid & 31;
    int hi = lane >> 4, ln = lane & 15;
    int m = wave & 3;                           // M-tile: rows 16m..16m+15
    const v8u* ap = (const v8u*)pa + (long)(m * KSTEPS) * 32 + lane;
    float pra = prelu[0];

    if (wave < 4) {
        // tiles j = 0, 1 (cols 0..31, always valid)
        v8f a0 = {}; v8f a1 = {};
        for (int s = 0; s < KSTEPS; ++s) {
            v16bf A = __builtin_bit_cast(v16bf, ap[(long)s * 32]);
            const unsigned short* bbase = &Xs[(long)s * 32 + hi * 16];
            const uint4* p0 = (const uint4*)(bbase + (0 * 16 + ln) * ROWPITCH);
            const uint4* p1 = (const uint4*)(bbase + (1 * 16 + ln) * ROWPITCH);
            v16bf B0 = mkbf16(p0[0], p0[1]);
            v16bf B1 = mkbf16(p1[0], p1[1]);
            a0 = __builtin_amdgcn_wmma_f32_16x16x32_bf16(false, A, false, B0, (short)0, a0, false, false);
            a1 = __builtin_amdgcn_wmma_f32_16x16x32_bf16(false, A, false, B1, (short)0, a1, false, false);
        }
#pragma unroll
        for (int j = 0; j < 2; ++j) {
            v8f acc = (j == 0) ? a0 : a1;
            int col = j * 16 + ln;              // < 32, always valid
            int n = nb + col;
            int kk = kid[col];
#pragma unroll
            for (int r = 0; r < 8; ++r) {
                int o = m * 16 + r + hi * 8;    // C/D layout: VGPR r -> row r+8*(lane/16)
                float v = acc[r] + bg[kk * OCH + o];
                v = (v >= 0.f) ? v : pra * v;
                out[(long)o * NSP + n] = v + xin[(long)o * NSP + n];
            }
        }
    } else {
        // tile j = 2 (cols 32..47; only 32..39 valid)
        v8f a2 = {};
        for (int s = 0; s < KSTEPS; ++s) {
            v16bf A = __builtin_bit_cast(v16bf, ap[(long)s * 32]);
            const unsigned short* bbase = &Xs[(long)s * 32 + hi * 16];
            const uint4* p2 = (const uint4*)(bbase + (2 * 16 + ln) * ROWPITCH);
            v16bf B2 = mkbf16(p2[0], p2[1]);
            a2 = __builtin_amdgcn_wmma_f32_16x16x32_bf16(false, A, false, B2, (short)0, a2, false, false);
        }
        int col = 32 + ln;
        if (col < DD) {
            int n = nb + col;
            int kk = kid[col];
#pragma unroll
            for (int r = 0; r < 8; ++r) {
                int o = m * 16 + r + hi * 8;
                float v = a2[r] + bg[kk * OCH + o];
                v = (v >= 0.f) ? v : pra * v;
                out[(long)o * NSP + n] = v + xin[(long)o * NSP + n];
            }
        }
    }
}

// ---------------- host-side orchestration ----------------
extern "C" void kernel_launch(void* const* d_in, const int* in_sizes, int n_in,
                              void* d_out, int out_size, void* d_ws, size_t ws_size,
                              hipStream_t stream) {
    const float* x      = (const float*)d_in[0];
    const float* proj_w = (const float*)d_in[1];
    const float* base_w = (const float*)d_in[2];
    const float* kw1 = (const float*)d_in[3];  const float* kb1 = (const float*)d_in[4];
    const float* kw2 = (const float*)d_in[5];  const float* kb2 = (const float*)d_in[6];
    const float* kw3 = (const float*)d_in[7];  const float* kb3 = (const float*)d_in[8];
    const float* bw1 = (const float*)d_in[9];  const float* bb1 = (const float*)d_in[10];
    const float* bw2 = (const float*)d_in[11]; const float* bb2 = (const float*)d_in[12];
    const float* bw3 = (const float*)d_in[13]; const float* bb3 = (const float*)d_in[14];
    const float* prelu = (const float*)d_in[15];
    float* out = (float*)d_out;

    char* ws = (char*)d_ws;
    float*          xp   = (float*)(ws + 0);                  // 64*64000 f32
    float*          pts  = (float*)(ws + 16384000);           // 64*64000 f32
    float*          cent = (float*)(ws + 32768000);           // 4*64 f32
    float*          sums = (float*)(ws + 32769024);           // 4*64 f32
    float*          cnts = (float*)(ws + 32770048);           // 4 f32
    int*            idx  = (int*)  (ws + 32770304);           // 64000 i32
    float*          wi   = (float*)(ws + 33026304);           // 108 f32
    float*          bg   = (float*)(ws + 33026816);           // 256 f32
    unsigned short* pa   = (unsigned short*)(ws + 33027840);  // 4*54*32*16 bf16 (32B aligned)

    k_proj<<<(8 * NSP + 255) / 256, 256, 0, stream>>>(x, proj_w, xp);
    k_mean<<<(CCH * NSP + 255) / 256, 256, 0, stream>>>(xp, pts);
    k_initcent<<<1, 256, 0, stream>>>(pts, cent);
    for (int it = 0; it < 10; ++it) {
        k_assign<<<NSP / 256, 256, 0, stream>>>(pts, cent, idx);
        k_reduce<<<NUMK * CCH, 256, 0, stream>>>(pts, idx, sums, cnts);
        k_update<<<1, 256, 0, stream>>>(sums, cnts, cent);
    }
    k_assign<<<NSP / 256, 256, 0, stream>>>(pts, cent, idx);   // final assignment
    k_mlp<<<1, 256, 0, stream>>>(cent, kw1, kb1, kw2, kb2, kw3, kb3,
                                 bw1, bb1, bw2, bb2, bw3, bb3, wi, bg);
    k_packA<<<(4 * KSTEPS * 512 + 255) / 256, 256, 0, stream>>>(base_w, pa);
    k_gemm<<<DD * DD, 256, NT * ROWPITCH * sizeof(unsigned short), stream>>>(
        xp, pa, idx, wi, bg, x, prelu, out);
}
